// Net_91122026152385
// MI455X (gfx1250) — matrix operand order repaired and checked by reading, not compiled, not run on hardware
//
#include <hip/hip_runtime.h>
#include <hip/hip_bf16.h>

// ---------------------------------------------------------------------------
// Types for CDNA5 WMMA (wave32): A/B = 16 bf16 per lane, C/D = 8 f32 per lane
// ---------------------------------------------------------------------------
typedef __bf16 v16bf __attribute__((ext_vector_type(16)));
typedef __bf16 v8bf  __attribute__((ext_vector_type(8)));
typedef float  v8f   __attribute__((ext_vector_type(8)));
typedef float  v4f   __attribute__((ext_vector_type(4)));

#define SHUF16(lo, hi) __builtin_shufflevector(lo, hi, 0,1,2,3,4,5,6,7,8,9,10,11,12,13,14,15)

// ---------------------------------------------------------------------------
// Generic fused-gather GEMM:  out[M,Nc] = act( A @ W + bias )
// A row i = concat( s0[i0?i0[i]:i, 0:k0], s1[i1?i1[i]:i, 0:k1], s2[i, 0:k2] )
// W is row-major [K, Nc] f32; operands converted to bf16 in LDS.
// Block tile 128x128, BK=32, 256 threads = 8 waves (2 x 4), each wave owns a
// 64x32 region = 4x2 v_wmma_f32_16x16x32_bf16 per k-step.
// LDS: A row-major [row][k], B transposed [col][k] so every fragment is two
// contiguous 16-byte runs per lane (2x ds_load_b128).
// ALIGNED=true: K%32==0, segment boundaries 16-aligned, row strides %4==0 ->
// no straddle, no per-element guards, unconditional float4 global loads.
// ---------------------------------------------------------------------------
template <bool ALIGNED>
__global__ __launch_bounds__(256) void gemm_wmma(
    int M, int Nc, int K,
    const float* __restrict__ s0, int k0, const int* __restrict__ i0,
    const float* __restrict__ s1, int k1, const int* __restrict__ i1,
    const float* __restrict__ s2, int k2, const int* __restrict__ i2,
    const float* __restrict__ W, const float* __restrict__ bias,
    float* __restrict__ out, int act)
{
    // row stride 40 elems = 80 B = 5x16B: keeps b128 alignment, spreads banks
    __shared__ __align__(16) __bf16 As[128][40];   // As[row][k]
    __shared__ __align__(16) __bf16 Bt[128][40];   // Bt[col][k]  (transposed)

    const int tid   = threadIdx.x;
    const int lane  = tid & 31;
    const int wv    = tid >> 5;          // 0..7
    const int waveM = wv >> 2;           // 0..1  (64-row slab)
    const int waveN = wv & 3;            // 0..3  (32-col slab)
    const int half  = lane >> 4;         // 0/1
    const int l15   = lane & 15;

    const int rowBase = blockIdx.y * 128;
    const int colBase = blockIdx.x * 128;

    // A-tile staging assignment: thread -> (row, 16-wide k run)
    const int ar = tid & 127;
    const int ac = (tid >> 7) * 16;      // 0 or 16
    const int gr = rowBase + ar;
    const bool rv = (gr < M);
    long b0 = 0, b1 = 0, b2 = 0;
    if (rv) {
        if (k0) b0 = (long)(i0 ? i0[gr] : gr) * (long)k0;
        if (k1) b1 = (long)(i1 ? i1[gr] : gr) * (long)k1;
        if (k2) b2 = (long)(i2 ? i2[gr] : gr) * (long)k2;
    }

    v8f acc[4][2];
    #pragma unroll
    for (int mt = 0; mt < 4; ++mt)
        #pragma unroll
        for (int nt = 0; nt < 2; ++nt)
            #pragma unroll
            for (int r = 0; r < 8; ++r)
                acc[mt][nt][r] = 0.0f;

    for (int kb = 0; kb < K; kb += 32) {
        // ---- stage A tile: gather + concat, f32->bf16, 2x ds_store_b128 ----
        {
            const int gkb = kb + ac;
            v8bf p0, p1;
            if (ALIGNED) {
                // run never straddles a segment; pointers 16B aligned
                if (rv) {
                    const float* sp;
                    if (gkb < k0)            sp = s0 + b0 + gkb;
                    else if (gkb < k0 + k1)  sp = s1 + b1 + (gkb - k0);
                    else                     sp = s2 + b2 + (gkb - k0 - k1);
                    const v4f* vp = (const v4f*)sp;
                    v4f f0 = vp[0], f1 = vp[1], f2 = vp[2], f3 = vp[3];
                    #pragma unroll
                    for (int i = 0; i < 4; ++i) {
                        p0[i]     = (__bf16)f0[i];
                        p0[4 + i] = (__bf16)f1[i];
                        p1[i]     = (__bf16)f2[i];
                        p1[4 + i] = (__bf16)f3[i];
                    }
                } else {
                    #pragma unroll
                    for (int i = 0; i < 8; ++i) { p0[i] = (__bf16)0.0f; p1[i] = (__bf16)0.0f; }
                }
            } else {
                const float* srcp = nullptr;
                if (rv) {
                    if (gkb + 15 < k0)                        srcp = s0 + b0 + gkb;
                    else if (gkb >= k0 && gkb + 15 < k0 + k1) srcp = s1 + b1 + (gkb - k0);
                    else if (gkb >= k0 + k1 && gkb + 15 < K)  srcp = s2 + b2 + (gkb - k0 - k1);
                }
                if (srcp && ((((unsigned long)srcp) & 15) == 0)) {
                    const v4f* vp = (const v4f*)srcp;
                    v4f f0 = vp[0], f1 = vp[1], f2 = vp[2], f3 = vp[3];
                    #pragma unroll
                    for (int i = 0; i < 4; ++i) {
                        p0[i]     = (__bf16)f0[i];
                        p0[4 + i] = (__bf16)f1[i];
                        p1[i]     = (__bf16)f2[i];
                        p1[4 + i] = (__bf16)f3[i];
                    }
                } else if (srcp) {
                    #pragma unroll
                    for (int i = 0; i < 8; ++i) {
                        p0[i] = (__bf16)srcp[i];
                        p1[i] = (__bf16)srcp[8 + i];
                    }
                } else {
                    #pragma unroll
                    for (int i = 0; i < 8; ++i) {
                        const int ga = gkb + i, gb = gkb + 8 + i;
                        float va = 0.0f, vb = 0.0f;
                        if (rv) {
                            if (ga < k0)           va = s0[b0 + ga];
                            else if (ga < k0 + k1) va = s1[b1 + ga - k0];
                            else if (ga < K)       va = s2[b2 + ga - k0 - k1];
                            if (gb < k0)           vb = s0[b0 + gb];
                            else if (gb < k0 + k1) vb = s1[b1 + gb - k0];
                            else if (gb < K)       vb = s2[b2 + gb - k0 - k1];
                        }
                        p0[i] = (__bf16)va;
                        p1[i] = (__bf16)vb;
                    }
                }
            }
            *(v8bf*)&As[ar][ac]     = p0;
            *(v8bf*)&As[ar][ac + 8] = p1;
        }
        // ---- stage B tile transposed: Bt[col][k], 1x ds_store_b128/unit ----
        #pragma unroll
        for (int u = 0; u < 2; ++u) {
            const int uu = tid + u * 256;
            const int j  = uu & 127;           // tile column
            const int ks = (uu >> 7) * 8;      // 0,8,16,24
            const int gn = colBase + j;
            const int gk0 = kb + ks;
            v8bf pk;
            if (ALIGNED) {
                if (gn < Nc) {
                    const float* wp = W + (long)gk0 * (long)Nc + gn;
                    #pragma unroll
                    for (int i = 0; i < 8; ++i) pk[i] = (__bf16)wp[(long)i * Nc];
                } else {
                    #pragma unroll
                    for (int i = 0; i < 8; ++i) pk[i] = (__bf16)0.0f;
                }
            } else {
                if (gn < Nc && gk0 + 7 < K) {
                    const float* wp = W + (long)gk0 * (long)Nc + gn;
                    #pragma unroll
                    for (int i = 0; i < 8; ++i) pk[i] = (__bf16)wp[(long)i * Nc];
                } else {
                    #pragma unroll
                    for (int i = 0; i < 8; ++i) {
                        const int gk = gk0 + i;
                        float v = (gk < K && gn < Nc) ? W[(long)gk * Nc + gn] : 0.0f;
                        pk[i] = (__bf16)v;
                    }
                }
            }
            *(v8bf*)&Bt[j][ks] = pk;
        }
        // prefetch next weight tile into caches (global_prefetch_b8)
        if (kb + 32 < K)
            __builtin_prefetch(W + (long)(kb + 32 + (tid >> 3)) * (long)Nc + colBase + (tid & 7) * 16, 0, 1);
        __syncthreads();

        // ---- fragments: 2x b128 per fragment, contiguous k order, same for A and B ----
        // B 32x16: lane n = l15, element e -> k = half*16 + e (contig in Bt)
        v16bf bfrag[2];
        #pragma unroll
        for (int nt = 0; nt < 2; ++nt) {
            const int j = waveN * 32 + nt * 16 + l15;
            v8bf lo = *(const v8bf*)&Bt[j][half * 16];
            v8bf hi = *(const v8bf*)&Bt[j][half * 16 + 8];
            bfrag[nt] = SHUF16(lo, hi);
        }
        // A 16x32: lane m = l15, element e -> k = half*16 + e (contig in As)
        v16bf afrag[4];
        #pragma unroll
        for (int mt = 0; mt < 4; ++mt) {
            const int rb = waveM * 64 + mt * 16 + l15;
            v8bf lo = *(const v8bf*)&As[rb][half * 16];
            v8bf hi = *(const v8bf*)&As[rb][half * 16 + 8];
            afrag[mt] = SHUF16(lo, hi);
        }
        // ---- 8 WMMAs ----
        #pragma unroll
        for (int mt = 0; mt < 4; ++mt)
            #pragma unroll
            for (int nt = 0; nt < 2; ++nt)
                acc[mt][nt] = __builtin_amdgcn_wmma_f32_16x16x32_bf16(
                    false, afrag[mt], false, bfrag[nt],
                    (short)0, acc[mt][nt], false, false);
        __syncthreads();
    }

    // ---- epilogue: bias + optional ELU, C/D layout (row=VGPR idx + 8*half) ----
    #pragma unroll
    for (int nt = 0; nt < 2; ++nt) {
        const int colg = colBase + waveN * 32 + nt * 16 + l15;
        if (colg >= Nc) continue;
        const float bv = bias ? bias[colg] : 0.0f;
        #pragma unroll
        for (int mt = 0; mt < 4; ++mt) {
            #pragma unroll
            for (int r = 0; r < 8; ++r) {
                const int rowg = rowBase + waveM * 64 + mt * 16 + half * 8 + r;
                if (rowg < M) {
                    float v = acc[mt][nt][r] + bv;
                    if (act) v = (v > 0.0f) ? v : (__expf(v) - 1.0f);
                    out[(long)rowg * (long)Nc + colg] = v;
                }
            }
        }
    }
}

// ---------------------------------------------------------------------------
// BatchNorm over axis 0 (per feature), one block per feature.
// ---------------------------------------------------------------------------
__global__ __launch_bounds__(256) void bn_kernel(
    const float* __restrict__ x, float* __restrict__ y,
    const float* __restrict__ gamma, const float* __restrict__ beta,
    int n, int f)
{
    const int feat = blockIdx.x;
    __shared__ float s1[256], s2[256];
    float a = 0.0f, b = 0.0f;
    for (int i = threadIdx.x; i < n; i += 256) {
        float v = x[(long)i * f + feat];
        a += v; b += v * v;
    }
    s1[threadIdx.x] = a; s2[threadIdx.x] = b;
    __syncthreads();
    for (int st = 128; st > 0; st >>= 1) {
        if (threadIdx.x < st) { s1[threadIdx.x] += s1[threadIdx.x + st];
                                s2[threadIdx.x] += s2[threadIdx.x + st]; }
        __syncthreads();
    }
    const float mu  = s1[0] / (float)n;
    const float var = s2[0] / (float)n - mu * mu;
    const float sc  = rsqrtf(var + 1e-5f) * gamma[feat];
    const float sh  = beta[feat];
    for (int i = threadIdx.x; i < n; i += 256) {
        const long o = (long)i * f + feat;
        y[o] = (x[o] - mu) * sc + sh;
    }
}

__global__ void zero_kernel(float* __restrict__ p, long n) {
    long i = (long)blockIdx.x * blockDim.x + threadIdx.x;
    long st = (long)gridDim.x * blockDim.x;
    for (; i < n; i += st) p[i] = 0.0f;
}

// one thread per (edge, 4 features): float4 read + 4 atomics
__global__ void scatter_add_kernel(const float* __restrict__ v, const int* __restrict__ idx,
                                   float* __restrict__ sums, float* __restrict__ cnt,
                                   long m, int c) {
    const int c4 = c >> 2;
    long t = (long)blockIdx.x * blockDim.x + threadIdx.x;
    if (t >= m * (long)c4) return;
    long e = t / c4; int f4 = (int)(t - e * c4);
    const int d = idx[e];
    const v4f val = *(const v4f*)(v + e * (long)c + f4 * 4);
    float* dst = sums + (long)d * c + f4 * 4;
    atomicAdd(dst + 0, val[0]);
    atomicAdd(dst + 1, val[1]);
    atomicAdd(dst + 2, val[2]);
    atomicAdd(dst + 3, val[3]);
    if (f4 == 0) atomicAdd(&cnt[d], 1.0f);
}

__global__ void divide_kernel(float* __restrict__ sums, const float* __restrict__ cnt,
                              long rows, int c) {
    long t = (long)blockIdx.x * blockDim.x + threadIdx.x;
    if (t >= rows * (long)c) return;
    sums[t] /= fmaxf(cnt[t / c], 1.0f);
}

__global__ void copy_kernel(const float* __restrict__ a, float* __restrict__ b, long n) {
    long i = (long)blockIdx.x * blockDim.x + threadIdx.x;
    if (i < n) b[i] = a[i];
}

// ---------------------------------------------------------------------------
// Host orchestration
// ---------------------------------------------------------------------------
static void gemm(hipStream_t st, long M, int Nc,
                 const float* s0, int k0, const int* i0,
                 const float* s1, int k1, const int* i1,
                 const float* s2, int k2, const int* i2,
                 const float* W, const float* bias, float* out, int act)
{
    const int K = k0 + k1 + k2;
    dim3 g((Nc + 127) / 128, (unsigned)((M + 127) / 128));
    const bool aligned = (K % 32 == 0) && (k0 % 16 == 0) &&
                         ((k0 + k1) % 16 == 0) && (k1 % 4 == 0) && (k2 % 4 == 0);
    if (aligned)
        gemm_wmma<true><<<g, 256, 0, st>>>((int)M, Nc, K, s0, k0, i0, s1, k1, i1,
                                           s2, k2, i2, W, bias, out, act);
    else
        gemm_wmma<false><<<g, 256, 0, st>>>((int)M, Nc, K, s0, k0, i0, s1, k1, i1,
                                            s2, k2, i2, W, bias, out, act);
}

static void zero(hipStream_t st, float* p, long n) {
    int blocks = (int)((n + 255) / 256); if (blocks > 4096) blocks = 4096;
    zero_kernel<<<blocks, 256, 0, st>>>(p, n);
}

extern "C" void kernel_launch(void* const* d_in, const int* in_sizes, int n_in,
                              void* d_out, int out_size, void* d_ws, size_t ws_size,
                              hipStream_t stream) {
    (void)n_in; (void)out_size; (void)ws_size;
    const float* x_in = (const float*)d_in[0];
    const float* e_in = (const float*)d_in[1];
    const int*   ei   = (const int*)d_in[2];
    const int*   nbat = (const int*)d_in[3];
    const int  N = in_sizes[0] / 10;
    const long E = in_sizes[1] / 5;
    const int  B = 64;
    const int* row = ei;
    const int* col = ei + E;

    auto f = [&](int i) { return (const float*)d_in[i]; };
    int pi = 4;  // params leaves, jax tree-flatten (sorted-key) order
    const float* bn_e_b = f(pi++); const float* bn_e_g = f(pi++);
    const float* bn_n_b = f(pi++); const float* bn_n_g = f(pi++);
    const float* lin1_b = f(pi++); const float* lin1_w = f(pi++);
    const float* lin2_b = f(pi++); const float* lin2_w = f(pi++);
    struct MetaP { const float *e1b,*e1w,*e2b,*e2w,*g1b,*g1w,*g2b,*g2w,
                   *n1ab,*n1aw,*n1bb,*n1bw,*n2ab,*n2aw,*n2bb,*n2bw; };
    MetaP mp[6];
    for (int m = 0; m < 6; ++m) {
        mp[m].e1b  = f(pi++); mp[m].e1w  = f(pi++);
        mp[m].e2b  = f(pi++); mp[m].e2w  = f(pi++);
        mp[m].g1b  = f(pi++); mp[m].g1w  = f(pi++);
        mp[m].g2b  = f(pi++); mp[m].g2w  = f(pi++);
        mp[m].n1ab = f(pi++); mp[m].n1aw = f(pi++);
        mp[m].n1bb = f(pi++); mp[m].n1bw = f(pi++);
        mp[m].n2ab = f(pi++); mp[m].n2aw = f(pi++);
        mp[m].n2bb = f(pi++); mp[m].n2bw = f(pi++);
    }

    // ---- workspace carve (floats) ----
    float* ws = (float*)d_ws; long off = 0;
    auto alloc = [&](long n) { float* p = ws + off; off += n; return p; };
    float* e0   = alloc(E * 5);        // bn'd edge_attr
    float* ebuf = alloc(E * 512);      // e activations (in-place across layers)
    float* t1   = alloc(E * 256);      // edge-sized temp A
    float* t2   = alloc(E * 256);      // edge-sized temp B
    float* xa   = alloc((long)N * 128);
    float* xb   = alloc((long)N * 128);
    float* ntmp = alloc((long)N * 256);
    float* agg  = alloc((long)N * 256);
    float* cntN = alloc(N);
    float* gsum = alloc((long)B * 128);
    float* cntB = alloc(B);
    float* gt   = alloc((long)B * 256);
    float* ua   = alloc((long)B * 256);
    float* ub   = alloc((long)B * 256);
    float* ybuf = alloc((long)B * 512);

    // ---- batchnorm ----
    bn_kernel<<<10, 256, 0, stream>>>(x_in, xa, bn_n_g, bn_n_b, N, 10);
    bn_kernel<<<5,  256, 0, stream>>>(e_in, e0, bn_e_g, bn_e_b, (int)E, 5);

    const int eout_[6] = {512, 512, 512, 512, 512, 128};
    const int gu_[6]   = {0, 128, 128, 128, 128, 128};
    const int gout_[6] = {128, 128, 128, 128, 128, 256};

    float* xcur = xa; float* xnext = xb;
    float* ucur = nullptr; float* unext = ua;
    const float* ecur = e0;
    int ef = 5, nf = 10;

    for (int m = 0; m < 6; ++m) {
        const MetaP& p = mp[m];
        const int e_out = eout_[m], n_out = 128, gu = gu_[m], g_out = gout_[m];

        // EdgeModel: e = MLP([x[row], x[col], e])
        gemm(stream, E, 128, xcur, nf, row, xcur, nf, col, ecur, ef, nullptr,
             p.e1w, p.e1b, t1, 1);
        gemm(stream, E, e_out, t1, 128, nullptr, nullptr, 0, nullptr, nullptr, 0, nullptr,
             p.e2w, p.e2b, ebuf, 0);

        // NodeModel message: m = MLP([x[row], e])
        gemm(stream, E, 256, xcur, nf, row, ebuf, e_out, nullptr, nullptr, 0, nullptr,
             p.n1aw, p.n1ab, t2, 1);
        gemm(stream, E, 256, t2, 256, nullptr, nullptr, 0, nullptr, nullptr, 0, nullptr,
             p.n1bw, p.n1bb, t1, 0);

        // scatter-mean messages -> dst nodes
        zero(stream, agg, (long)N * 256);
        zero(stream, cntN, N);
        scatter_add_kernel<<<(int)((E * 64 + 255) / 256), 256, 0, stream>>>(
            t1, col, agg, cntN, E, 256);
        divide_kernel<<<(int)(((long)N * 256 + 255) / 256), 256, 0, stream>>>(
            agg, cntN, N, 256);

        // NodeModel update: x = MLP([x, agg])
        gemm(stream, N, 256, xcur, nf, nullptr, agg, 256, nullptr, nullptr, 0, nullptr,
             p.n2aw, p.n2ab, ntmp, 1);
        gemm(stream, N, n_out, ntmp, 256, nullptr, nullptr, 0, nullptr, nullptr, 0, nullptr,
             p.n2bw, p.n2bb, xnext, 0);

        // GlobalModel: g = seg_mean(x, batch); u = MLP([u, g])
        zero(stream, gsum, (long)B * n_out);
        zero(stream, cntB, B);
        scatter_add_kernel<<<(int)(((long)N * (n_out / 4) + 255) / 256), 256, 0, stream>>>(
            xnext, nbat, gsum, cntB, N, n_out);
        divide_kernel<<<(int)(((long)B * n_out + 255) / 256), 256, 0, stream>>>(
            gsum, cntB, B, n_out);
        if (gu)
            gemm(stream, B, 256, ucur, gu, nullptr, gsum, n_out, nullptr, nullptr, 0, nullptr,
                 p.g1w, p.g1b, gt, 1);
        else
            gemm(stream, B, 256, gsum, n_out, nullptr, nullptr, 0, nullptr, nullptr, 0, nullptr,
                 p.g1w, p.g1b, gt, 1);
        gemm(stream, B, g_out, gt, 256, nullptr, nullptr, 0, nullptr, nullptr, 0, nullptr,
             p.g2w, p.g2b, unext, 0);

        // rotate
        ecur = ebuf; ef = e_out;
        float* tx = xcur; xcur = xnext; xnext = tx;
        nf = n_out;
        ucur = unext; unext = (ucur == ua) ? ub : ua;
    }

    // head: out = (u, lin2(elu(lin1(u))))
    float* outp = (float*)d_out;
    copy_kernel<<<(int)(((long)B * 256 + 255) / 256), 256, 0, stream>>>(ucur, outp, (long)B * 256);
    gemm(stream, B, 512, ucur, 256, nullptr, nullptr, 0, nullptr, nullptr, 0, nullptr,
         lin1_w, lin1_b, ybuf, 1);
    gemm(stream, B, 14541, ybuf, 512, nullptr, nullptr, 0, nullptr, nullptr, 0, nullptr,
         lin2_w, lin2_b, outp + (long)B * 256, 0);
}
